// PrefillChunk1_51891794870755
// MI455X (gfx1250) — matrix operand order repaired and checked by reading, not compiled, not run on hardware
//
#include <hip/hip_runtime.h>
#include <hip/hip_bf16.h>
#include <math.h>

typedef __attribute__((ext_vector_type(16))) __bf16 v16bf;
typedef __attribute__((ext_vector_type(8)))  float  v8f;

#define N_SEQ 1024
#define HID   2048
#define NH    8
#define NKV   2
#define NREP  4
#define HD    256
#define DFF   4096
#define PLD   256
#define NLT   30
#define NLAYERS 8

// ---------------------------------------------------------------------------
// reductions
// ---------------------------------------------------------------------------
__device__ __forceinline__ float wave_sum(float v) {
#pragma unroll
  for (int o = 16; o > 0; o >>= 1) v += __shfl_down(v, o, 32);
  return v;
}
__device__ __forceinline__ float wave_max(float v) {
#pragma unroll
  for (int o = 16; o > 0; o >>= 1) v = fmaxf(v, __shfl_down(v, o, 32));
  return v;
}
__device__ __forceinline__ float block_sum(float v, float* sm) {
  __syncthreads();
  int t = threadIdx.x;
  v = wave_sum(v);
  if ((t & 31) == 0) sm[t >> 5] = v;
  __syncthreads();
  int nw = blockDim.x >> 5;
  float r = (t < nw) ? sm[t] : 0.0f;
  r = wave_sum(r);
  if (t == 0) sm[0] = r;
  __syncthreads();
  return sm[0];
}
__device__ __forceinline__ float block_max(float v, float* sm) {
  __syncthreads();
  int t = threadIdx.x;
  v = wave_max(v);
  if ((t & 31) == 0) sm[t >> 5] = v;
  __syncthreads();
  int nw = blockDim.x >> 5;
  float r = (t < nw) ? sm[t] : -3.4e38f;
  r = wave_max(r);
  if (t == 0) sm[0] = r;
  __syncthreads();
  return sm[0];
}
__device__ __forceinline__ float gelu_t(float x) {
  float x3 = x * x * x;
  return 0.5f * x * (1.0f + tanhf(0.79788456080286536f * (x + 0.044715f * x3)));
}

// ---------------------------------------------------------------------------
// Generic batched GEMM:  C[m,n] = alpha * sum_k A[m,k] * W[n,k]
// A: [M,K] row-major (lda), W: [N,K] row-major (ldw), C: row-major (ldc).
// Batch b: A += b*sA, W += (b/wdiv)*sW, C += b*sC.
// One wave computes a 32x32 C tile; bf16x3 split gives ~fp32 accuracy on
// v_wmma_f32_16x16x32_bf16 (12 WMMA per 32-wide K step).
// ---------------------------------------------------------------------------
#define WMMA_BF16(A_, B_, C_) \
  __builtin_amdgcn_wmma_f32_16x16x32_bf16(false, (A_), false, (B_), (short)0, (C_), false, false)

__global__ __launch_bounds__(128) void k_gemm(
    const float* __restrict__ A, const float* __restrict__ W, float* __restrict__ C,
    int M, int N, int K, int lda, int ldw, int ldc, float alpha,
    long long sA, long long sW, long long sC, int wdiv)
{
  const int b = blockIdx.y;
  A += (long long)b * sA;
  W += (long long)(b / wdiv) * sW;
  C += (long long)b * sC;

  const int lane = threadIdx.x & 31;
  const int wave = threadIdx.x >> 5;
  const int tN = N >> 5;
  const int tile = blockIdx.x * 4 + wave;
  const int ntiles = (M >> 5) * tN;
  if (tile >= ntiles) return;
  const int tm = (tile / tN) << 5;
  const int tn = (tile % tN) << 5;

  const int r   = lane & 15;
  const int kbA = (lane >> 4) << 4;   // A frag: K base 0 or 16 (contiguous 16)
  const int kbB = (lane >> 4) << 4;   // B frag: K base 0 or 16 (contiguous 16)

  const float* a0 = A + (long long)(tm + r) * lda;
  const float* a1 = a0 + (long long)16 * lda;
  const float* w0 = W + (long long)(tn + r) * ldw;
  const float* w1 = w0 + (long long)16 * ldw;

  v8f acc00 = {}, acc01 = {}, acc10 = {}, acc11 = {};

  for (int k0 = 0; k0 < K; k0 += 32) {
    float af0[16], af1[16], bf0[16], bf1[16];
    const float4* pa0 = (const float4*)(a0 + k0 + kbA);
    const float4* pa1 = (const float4*)(a1 + k0 + kbA);
    const float4* pb0 = (const float4*)(w0 + k0 + kbB);
    const float4* pb1 = (const float4*)(w1 + k0 + kbB);
    *(float4*)&af0[0]  = pa0[0]; *(float4*)&af0[4]  = pa0[1];
    *(float4*)&af0[8]  = pa0[2]; *(float4*)&af0[12] = pa0[3];
    *(float4*)&af1[0]  = pa1[0]; *(float4*)&af1[4]  = pa1[1];
    *(float4*)&af1[8]  = pa1[2]; *(float4*)&af1[12] = pa1[3];
    *(float4*)&bf0[0]  = pb0[0]; *(float4*)&bf0[4]  = pb0[1];
    *(float4*)&bf0[8]  = pb0[2]; *(float4*)&bf0[12] = pb0[3];
    *(float4*)&bf1[0]  = pb1[0]; *(float4*)&bf1[4]  = pb1[1];
    *(float4*)&bf1[8]  = pb1[2]; *(float4*)&bf1[12] = pb1[3];

    v16bf a0h, a0l, a1h, a1l, b0h, b0l, b1h, b1l;
#pragma unroll
    for (int i = 0; i < 16; ++i) {
      __bf16 h;
      h = (__bf16)af0[i]; a0h[i] = h; a0l[i] = (__bf16)(af0[i] - (float)h);
      h = (__bf16)af1[i]; a1h[i] = h; a1l[i] = (__bf16)(af1[i] - (float)h);
      h = (__bf16)bf0[i]; b0h[i] = h; b0l[i] = (__bf16)(bf0[i] - (float)h);
      h = (__bf16)bf1[i]; b1h[i] = h; b1l[i] = (__bf16)(bf1[i] - (float)h);
    }
    acc00 = WMMA_BF16(a0h, b0h, acc00);
    acc00 = WMMA_BF16(a0h, b0l, acc00);
    acc00 = WMMA_BF16(a0l, b0h, acc00);
    acc01 = WMMA_BF16(a0h, b1h, acc01);
    acc01 = WMMA_BF16(a0h, b1l, acc01);
    acc01 = WMMA_BF16(a0l, b1h, acc01);
    acc10 = WMMA_BF16(a1h, b0h, acc10);
    acc10 = WMMA_BF16(a1h, b0l, acc10);
    acc10 = WMMA_BF16(a1l, b0h, acc10);
    acc11 = WMMA_BF16(a1h, b1h, acc11);
    acc11 = WMMA_BF16(a1h, b1l, acc11);
    acc11 = WMMA_BF16(a1l, b1h, acc11);
  }

  const int crow = (lane >> 4) << 3;
  const int ccol = lane & 15;
#pragma unroll
  for (int j = 0; j < 8; ++j) {
    C[(long long)(tm + crow + j) * ldc + (tn + ccol)]            = acc00[j] * alpha;
    C[(long long)(tm + crow + j) * ldc + (tn + 16 + ccol)]       = acc01[j] * alpha;
    C[(long long)(tm + 16 + crow + j) * ldc + (tn + ccol)]       = acc10[j] * alpha;
    C[(long long)(tm + 16 + crow + j) * ldc + (tn + 16 + ccol)]  = acc11[j] * alpha;
  }
}

// ---------------------------------------------------------------------------
// elementwise / normalization kernels
// ---------------------------------------------------------------------------
__global__ void k_rms(const float* __restrict__ x, const float* __restrict__ w,
                      float* __restrict__ out, int D)
{
  __shared__ float sm[32];
  const long long row = blockIdx.x;
  const float* xr = x + row * D;
  float* orow = out + row * D;
  float ss = 0.0f;
  for (int i = threadIdx.x; i < D; i += blockDim.x) { float v = xr[i]; ss += v * v; }
  float tot = block_sum(ss, sm);
  float inv = rsqrtf(tot / (float)D + 1e-6f);
  for (int i = threadIdx.x; i < D; i += blockDim.x)
    orow[i] = xr[i] * inv * (1.0f + w[i]);
}

__global__ void k_add_rms(const float* __restrict__ res, const float* __restrict__ x,
                          const float* __restrict__ w, float* __restrict__ dst,
                          int D, const float* __restrict__ scl, int sidx)
{
  __shared__ float sm[32];
  const long long row = blockIdx.x;
  const float* xr = x + row * D;
  const float* rr = res + row * D;
  float* dr = dst + row * D;
  float ss = 0.0f;
  for (int i = threadIdx.x; i < D; i += blockDim.x) { float v = xr[i]; ss += v * v; }
  float tot = block_sum(ss, sm);
  float inv = rsqrtf(tot / (float)D + 1e-6f);
  float sc = scl ? scl[sidx] : 1.0f;
  for (int i = threadIdx.x; i < D; i += blockDim.x)
    dr[i] = (rr[i] + xr[i] * inv * (1.0f + w[i])) * sc;
}

// plc = (rms256(proj) + raw) * 2^-1/2 ; proj already scaled by PROJ_SCALE
__global__ void k_plc(const float* __restrict__ proj, const float* __restrict__ raw,
                      const float* __restrict__ w, float* __restrict__ plc)
{
  __shared__ float sm[32];
  const int n = blockIdx.x, t = blockIdx.y, d = threadIdx.x;
  const long long idx = (long long)n * (NLT * PLD) + (long long)t * PLD + d;
  float x = proj[idx];
  float tot = block_sum(x * x, sm);
  float y = x * rsqrtf(tot / (float)PLD + 1e-6f) * (1.0f + w[d]);
  plc[idx] = (y + raw[idx]) * 0.70710678118654752f;
}

// QK rms-norm + RoPE, V rms-norm; writes K/V into d_out and V^T for the AV GEMM
__global__ void k_qkv(const float* __restrict__ qraw, const float* __restrict__ kraw,
                      const float* __restrict__ vraw,
                      const float* __restrict__ wqn, const float* __restrict__ wkn,
                      const float* __restrict__ ct, const float* __restrict__ st,
                      float* __restrict__ qh, float* __restrict__ kout,
                      float* __restrict__ vout, float* __restrict__ vt)
{
  __shared__ float sm[32];
  __shared__ float buf[HD];
  const int n = blockIdx.x;
  const int d = threadIdx.x;
  const float c = ct[(long long)n * HD + d];
  const float s = st[(long long)n * HD + d];

  for (int h = 0; h < NH; ++h) {
    float x = qraw[(long long)n * (NH * HD) + h * HD + d];
    float tot = block_sum(x * x, sm);
    float y = x * rsqrtf(tot / (float)HD + 1e-6f) * (1.0f + wqn[d]);
    buf[d] = y;
    __syncthreads();
    float rot = (d < HD / 2) ? -buf[d + HD / 2] : buf[d - HD / 2];
    __syncthreads();
    qh[((long long)h * N_SEQ + n) * HD + d] = y * c + rot * s;
  }
  for (int h = 0; h < NKV; ++h) {
    float x = kraw[(long long)n * (NKV * HD) + h * HD + d];
    float tot = block_sum(x * x, sm);
    float y = x * rsqrtf(tot / (float)HD + 1e-6f) * (1.0f + wkn[d]);
    buf[d] = y;
    __syncthreads();
    float rot = (d < HD / 2) ? -buf[d + HD / 2] : buf[d - HD / 2];
    __syncthreads();
    kout[((long long)h * N_SEQ + n) * HD + d] = y * c + rot * s;

    float vx = vraw[(long long)n * (NKV * HD) + h * HD + d];
    float vtot = block_sum(vx * vx, sm);
    float vy = vx * rsqrtf(vtot / (float)HD + 1e-6f);
    vout[((long long)h * N_SEQ + n) * HD + d] = vy;
    vt[(long long)h * HD * N_SEQ + (long long)d * N_SEQ + n] = vy;
  }
}

__global__ void k_softmax(float* __restrict__ sc)
{
  __shared__ float sm[32];
  const int qrow = blockIdx.x, h = blockIdx.y;
  float* row = sc + ((long long)h * N_SEQ + qrow) * N_SEQ;
  const int t = threadIdx.x;
  float v[4];
  float mx = -3.4e38f;
#pragma unroll
  for (int j = 0; j < 4; ++j) {
    int c = t + j * 256;
    float x = row[c] + ((c > qrow) ? -1.0e9f : 0.0f);
    v[j] = x;
    mx = fmaxf(mx, x);
  }
  float gmax = block_max(mx, sm);
  float s = 0.0f;
#pragma unroll
  for (int j = 0; j < 4; ++j) { v[j] = __expf(v[j] - gmax); s += v[j]; }
  float gs = block_sum(s, sm);
  float inv = 1.0f / gs;
#pragma unroll
  for (int j = 0; j < 4; ++j) row[t + j * 256] = v[j] * inv;
}

__global__ void k_gelu_mul(float* __restrict__ g, const float* __restrict__ u, int n)
{
  int i = blockIdx.x * blockDim.x + threadIdx.x;
  if (i < n) g[i] = gelu_t(g[i]) * u[i];
}

__global__ void k_gelu_sl(float* __restrict__ g, const float* __restrict__ plc, int layer)
{
  int i = blockIdx.x * blockDim.x + threadIdx.x;  // N_SEQ*PLD
  int n = i >> 8, p = i & 255;
  g[i] = gelu_t(g[i]) * plc[(long long)n * (NLT * PLD) + layer * PLD + p];
}

// ---------------------------------------------------------------------------
// host
// ---------------------------------------------------------------------------
static void gemm(hipStream_t st, const float* A, const float* W, float* C,
                 int M, int N, int K, int lda, int ldw, int ldc, float alpha,
                 int batch = 1, long long sA = 0, long long sW = 0, long long sC = 0,
                 int wdiv = 1)
{
  int tiles = (M / 32) * (N / 32);
  dim3 g((tiles + 3) / 4, batch);
  k_gemm<<<g, 128, 0, st>>>(A, W, C, M, N, K, lda, ldw, ldc, alpha, sA, sW, sC, wdiv);
}

extern "C" void kernel_launch(void* const* d_in, const int* in_sizes, int n_in,
                              void* d_out, int out_size, void* d_ws, size_t ws_size,
                              hipStream_t stream)
{
  (void)in_sizes; (void)n_in; (void)out_size; (void)ws_size;

  const float* hs_in   = (const float*)d_in[0];
  /* d_in[1] = causal_mask (recomputed in-kernel) */
  const float* plraw   = (const float*)d_in[2];
  const float* cosS    = (const float*)d_in[3];
  const float* sinS    = (const float*)d_in[4];
  const float* cosF    = (const float*)d_in[5];
  const float* sinF    = (const float*)d_in[6];
  const float* w_in_ln = (const float*)d_in[7];
  const float* Wq      = (const float*)d_in[8];
  const float* w_qn    = (const float*)d_in[9];
  const float* Wk      = (const float*)d_in[10];
  const float* w_kn    = (const float*)d_in[11];
  const float* Wv      = (const float*)d_in[12];
  const float* Wo      = (const float*)d_in[13];
  const float* w_pa    = (const float*)d_in[14];
  const float* w_pf    = (const float*)d_in[15];
  const float* Wg      = (const float*)d_in[16];
  const float* Wu      = (const float*)d_in[17];
  const float* Wd      = (const float*)d_in[18];
  const float* w_pff   = (const float*)d_in[19];
  const float* Wplg    = (const float*)d_in[20];
  const float* Wplp    = (const float*)d_in[21];
  const float* w_ppl   = (const float*)d_in[22];
  const float* lsc     = (const float*)d_in[23];
  const float* Wplm    = (const float*)d_in[24];
  const float* w_pln   = (const float*)d_in[25];

  float* out     = (float*)d_out;
  float* out_hs  = out;
  float* out_plc = out + (long long)N_SEQ * HID;
  float* out_kv  = out_plc + (long long)N_SEQ * NLT * PLD;

  float* p   = (float*)d_ws;
  float* hs  = p; p += (long long)N_SEQ * HID;        // running hidden state
  float* hb  = p; p += (long long)N_SEQ * HID;        // rms'ed input
  float* t0  = p; p += (long long)N_SEQ * HID;        // q raw / proj outs
  float* krw = p; p += (long long)N_SEQ * NKV * HD;
  float* vrw = p; p += (long long)N_SEQ * NKV * HD;
  float* qh  = p; p += (long long)NH * N_SEQ * HD;    // roped Q, head-major
  float* vt  = p; p += (long long)NKV * HD * N_SEQ;   // V^T
  float* ao  = p; p += (long long)N_SEQ * NH * HD;    // attention out
  float* plg = p; p += (long long)N_SEQ * PLD;
  float* big = p;                                     // 8M floats: proj | scores | ffg+ffu
  float* ffg = big;
  float* ffu = big + (long long)N_SEQ * DFF;

  // --- per-layer input channel: proj = hs @ W_plm^T * HID^-0.5, rms256, +raw ---
  gemm(stream, hs_in, Wplm, big, N_SEQ, NLT * PLD, HID, HID, HID, NLT * PLD,
       0.022097086912079608f /* 2048^-0.5 */);
  { dim3 g(N_SEQ, NLT); k_plc<<<g, PLD, 0, stream>>>(big, plraw, w_pln, out_plc); }

  const float* cur = hs_in;
  for (int i = 0; i < NLAYERS; ++i) {
    const bool full = (i % 5 == 4);
    const float* cosT = full ? cosF : cosS;
    const float* sinT = full ? sinF : sinS;
    float* kout = out_kv + (long long)i * 2 * NKV * N_SEQ * HD;
    float* vout = kout + (long long)NKV * N_SEQ * HD;

    // --- attention ---
    k_rms<<<N_SEQ, 256, 0, stream>>>(cur, w_in_ln + (long long)i * HID, hb, HID);
    gemm(stream, hb, Wq + (long long)i * NH * HD * HID, t0, N_SEQ, NH * HD, HID, HID, HID, NH * HD, 1.0f);
    gemm(stream, hb, Wk + (long long)i * NKV * HD * HID, krw, N_SEQ, NKV * HD, HID, HID, HID, NKV * HD, 1.0f);
    gemm(stream, hb, Wv + (long long)i * NKV * HD * HID, vrw, N_SEQ, NKV * HD, HID, HID, HID, NKV * HD, 1.0f);
    k_qkv<<<N_SEQ, HD, 0, stream>>>(t0, krw, vrw, w_qn + (long long)i * HD, w_kn + (long long)i * HD,
                                    cosT, sinT, qh, kout, vout, vt);
    // scores[h] = Q[h] @ K[h/4]^T   (batched over 8 heads, GQA via wdiv=4)
    gemm(stream, qh, kout, big, N_SEQ, N_SEQ, HD, HD, HD, N_SEQ, 1.0f,
         NH, (long long)N_SEQ * HD, (long long)N_SEQ * HD, (long long)N_SEQ * N_SEQ, NREP);
    { dim3 g(N_SEQ, NH); k_softmax<<<g, 256, 0, stream>>>(big); }
    // ao[:, h*HD:(h+1)*HD] = P[h] @ V[h/4]   (B = V^T so K is contiguous)
    gemm(stream, big, vt, ao, N_SEQ, HD, N_SEQ, N_SEQ, N_SEQ, NH * HD, 1.0f,
         NH, (long long)N_SEQ * N_SEQ, (long long)HD * N_SEQ, (long long)HD, NREP);
    gemm(stream, ao, Wo + (long long)i * HID * NH * HD, t0, N_SEQ, HID, NH * HD, NH * HD, NH * HD, HID, 1.0f);
    k_add_rms<<<N_SEQ, 256, 0, stream>>>(cur, t0, w_pa + (long long)i * HID, hs, HID, nullptr, 0);
    cur = hs;

    // --- gated MLP ---
    k_rms<<<N_SEQ, 256, 0, stream>>>(hs, w_pf + (long long)i * HID, hb, HID);
    gemm(stream, hb, Wg + (long long)i * DFF * HID, ffg, N_SEQ, DFF, HID, HID, HID, DFF, 1.0f);
    gemm(stream, hb, Wu + (long long)i * DFF * HID, ffu, N_SEQ, DFF, HID, HID, HID, DFF, 1.0f);
    k_gelu_mul<<<(N_SEQ * DFF) / 256, 256, 0, stream>>>(ffg, ffu, N_SEQ * DFF);
    gemm(stream, ffg, Wd + (long long)i * HID * DFF, t0, N_SEQ, HID, DFF, DFF, DFF, HID, 1.0f);
    k_add_rms<<<N_SEQ, 256, 0, stream>>>(hs, t0, w_pff + (long long)i * HID, hs, HID, nullptr, 0);

    // --- per-layer side channel ---
    gemm(stream, hs, Wplg + (long long)i * PLD * HID, plg, N_SEQ, PLD, HID, HID, HID, PLD, 1.0f);
    k_gelu_sl<<<(N_SEQ * PLD) / 256, 256, 0, stream>>>(plg, out_plc, i);
    gemm(stream, plg, Wplp + (long long)i * HID * PLD, t0, N_SEQ, HID, PLD, PLD, PLD, HID, 1.0f);
    float* dst = (i == NLAYERS - 1) ? out_hs : hs;
    k_add_rms<<<N_SEQ, 256, 0, stream>>>(hs, t0, w_ppl + (long long)i * HID, dst, HID, lsc, i);
  }
}
